// MyLinearCrossEntropyLoss_79663053406262
// MI455X (gfx1250) — matrix operand order repaired and checked by reading, not compiled, not run on hardware
//
#include <hip/hip_runtime.h>
#include <hip/hip_bf16.h>

// ---------------------------------------------------------------------------
// Fused linear + weighted cross-entropy for MI455X (gfx1250, wave32, WMMA).
//
// Shapes: x[B=4096, D=2048] f32, L[V=32000, D=2048] f32, target[B] int,
//         class_weight[V] f32  ->  scalar mean loss.
//
//  (1) convert x, L  f32 -> bf16 in workspace (bf16 L = 131MB fits 192MB L2)
//  (2) tiled GEMM with v_wmma_f32_16x16x32_bf16 (64x64 wave tile ->
//      1.0 global_load_b128 per WMMA, feeding the XDL at sustainable
//      fetch bandwidth), fused online-softmax epilogue producing
//      per-(row, 64-col-chunk) (max, sumexp) partials + target-logit scatter
//  (3) per-row merge of partials -> logZ -> weighted NLL, atomic accumulate
//  (4) finalize: out = num / den
// ---------------------------------------------------------------------------

#define B_ROWS 4096
#define D_DIM  2048
#define V_DIM  32000
#define NCHUNK (V_DIM / 64)     // 500 column chunks of 64
#define IGNORE_INDEX (-100)

typedef __attribute__((ext_vector_type(16))) __bf16 v16bf;
typedef __attribute__((ext_vector_type(8)))  float  v8f;

union Frag { v16bf v; uint4 q[2]; };

// ------------------------------- pass 0: f32 -> bf16 -----------------------
__device__ __forceinline__ unsigned short f2bf(float f) {
  unsigned u = __float_as_uint(f);
  unsigned r = u + 0x7FFFu + ((u >> 16) & 1u);   // round-to-nearest-even
  return (unsigned short)(r >> 16);
}

__global__ __launch_bounds__(256) void cvt_f32_bf16(const float* __restrict__ in,
                                                    unsigned short* __restrict__ out,
                                                    int n4) {
  int i = blockIdx.x * blockDim.x + threadIdx.x;
  if (i >= n4) return;
  float4 f = ((const float4*)in)[i];
  ushort4 o;
  o.x = f2bf(f.x); o.y = f2bf(f.y); o.z = f2bf(f.z); o.w = f2bf(f.w);
  ((ushort4*)out)[i] = o;
}

// ------------------------------- init accumulators -------------------------
__global__ void init_accum(float* accum) {
  if (threadIdx.x < 2) accum[threadIdx.x] = 0.0f;
}

// ------------------------------- pass 1: GEMM + fused CE epilogue ----------
// block = 256 threads (8 waves), arranged 2 (M) x 4 (N).
// wave tile 64x64 (4x4 subtiles of 16x16) -> 16 b128 loads : 16 WMMA.
// block tile 128 (rows) x 256 (vocab cols). grid = (V/256, B/128).
__global__ __launch_bounds__(256) void gemm_ce(
    const __bf16* __restrict__ xb,    // [B, D] bf16
    const __bf16* __restrict__ Lb,    // [V, D] bf16
    const int*    __restrict__ tgt,   // [B]
    float2*       __restrict__ part,  // [B, NCHUNK] (max, sumexp)
    float*        __restrict__ gath)  // [B] logit at target column
{
  const int lane = threadIdx.x & 31;
  const int wave = threadIdx.x >> 5;
  const int wm   = wave & 1;          // M sub-offset: wm*64
  const int wn   = wave >> 1;         // N sub-offset: wn*64
  const int ln   = lane & 15;         // 0..15 within half-wave
  const int lg   = lane >> 4;         // 0 or 1 (half-wave group)

  const int m0 = blockIdx.y * 128 + wm * 64;   // first row of this wave
  const int v0 = blockIdx.x * 256 + wn * 64;   // first vocab col of this wave

  v8f acc[4][4];
#pragma unroll
  for (int i = 0; i < 4; ++i)
#pragma unroll
    for (int j = 0; j < 4; ++j) acc[i][j] = (v8f){0,0,0,0,0,0,0,0};

  // A fragment (16-bit A 16x32, ISA 7.12.2):
  //   lanes 0-15 : M=lane,    K chunks [0..7] and [16..23]
  //   lanes 16-31: M=lane-16, K chunks [8..15] and [24..31]
  const __bf16* aptr[4];
#pragma unroll
  for (int i = 0; i < 4; ++i)
    aptr[i] = xb + (size_t)(m0 + i * 16 + ln) * D_DIM + lg * 8;
  // B fragment (16-bit B 32x16): lanes 0-15 col=lane K chunks [0..7] and
  // [16..23], lanes 16-31 col=lane-16 K chunks [8..15] and [24..31].
  const __bf16* bptr[4];
#pragma unroll
  for (int j = 0; j < 4; ++j)
    bptr[j] = Lb + (size_t)(v0 + j * 16 + ln) * D_DIM + lg * 8;

  for (int k0 = 0; k0 < D_DIM; k0 += 32) {
    Frag a[4], b[4];
#pragma unroll
    for (int i = 0; i < 4; ++i) {
      a[i].q[0] = *(const uint4*)(aptr[i] + k0);
      a[i].q[1] = *(const uint4*)(aptr[i] + k0 + 16);
    }
#pragma unroll
    for (int j = 0; j < 4; ++j) {
      b[j].q[0] = *(const uint4*)(bptr[j] + k0);
      b[j].q[1] = *(const uint4*)(bptr[j] + k0 + 16);
    }
#pragma unroll
    for (int i = 0; i < 4; ++i)
#pragma unroll
      for (int j = 0; j < 4; ++j)
        acc[i][j] = __builtin_amdgcn_wmma_f32_16x16x32_bf16(
            false, a[i].v, false, b[j].v, (short)0, acc[i][j], false, false);
  }

  // Epilogue: per-row online softmax stats over this wave's 64 columns.
  // C layout: VGPR r, lanes 0-15 -> (M=r, N=lane); lanes 16-31 -> (M=8+r).
  // A row lives across one 16-lane group -> butterfly shuffles with masks
  // 1,2,4,8 (stay inside the group on wave32).
  const int chunk = blockIdx.x * 4 + wn;       // 64-col chunk index
#pragma unroll
  for (int i = 0; i < 4; ++i) {
#pragma unroll
    for (int r = 0; r < 8; ++r) {
      const int row = m0 + i * 16 + lg * 8 + r;
      float vmax = acc[i][0][r];
      vmax = fmaxf(vmax, acc[i][1][r]);
      vmax = fmaxf(vmax, acc[i][2][r]);
      vmax = fmaxf(vmax, acc[i][3][r]);
#pragma unroll
      for (int off = 1; off < 16; off <<= 1)
        vmax = fmaxf(vmax, __shfl_xor(vmax, off, 32));
      float s = __expf(acc[i][0][r] - vmax) + __expf(acc[i][1][r] - vmax)
              + __expf(acc[i][2][r] - vmax) + __expf(acc[i][3][r] - vmax);
#pragma unroll
      for (int off = 1; off < 16; off <<= 1)
        s += __shfl_xor(s, off, 32);

      // scatter target logit (exactly one (block,lane,j) matches per row)
      const int t = tgt[row];
#pragma unroll
      for (int j = 0; j < 4; ++j) {
        const int col = v0 + j * 16 + ln;
        if (col == t) gath[row] = acc[i][j][r];
      }
      if (ln == 0) part[(size_t)row * NCHUNK + chunk] = make_float2(vmax, s);
    }
  }
}

// ------------------------------- pass 2: per-row merge + weighted NLL ------
// one wave per row; 8 rows per 256-thread block.
__global__ __launch_bounds__(256) void reduce_rows(
    const float2* __restrict__ part, const float* __restrict__ gath,
    const int* __restrict__ tgt, const float* __restrict__ cw,
    float* __restrict__ accum)   // accum[0]=sum(w*(logZ-g)), accum[1]=sum(w)
{
  const int lane = threadIdx.x & 31;
  const int wave = threadIdx.x >> 5;
  const int row  = blockIdx.x * 8 + wave;
  if (row >= B_ROWS) return;

  float m = -3.4e38f, s = 0.0f;
  for (int c = lane; c < NCHUNK; c += 32) {
    float2 p = part[(size_t)row * NCHUNK + c];
    float nm = fmaxf(m, p.x);
    s = s * __expf(m - nm) + p.y * __expf(p.x - nm);
    m = nm;
  }
#pragma unroll
  for (int off = 1; off < 32; off <<= 1) {
    float om = __shfl_xor(m, off, 32);
    float os = __shfl_xor(s, off, 32);
    float nm = fmaxf(m, om);
    s = s * __expf(m - nm) + os * __expf(om - nm);
    m = nm;
  }
  if (lane == 0) {
    const float logZ = m + __logf(s);
    const int t = tgt[row];
    const bool valid = (t != IGNORE_INDEX);
    const float w = valid ? cw[t] : 0.0f;
    const float g = valid ? gath[row] : 0.0f;
    atomicAdd(&accum[0], w * (logZ - g));
    atomicAdd(&accum[1], w);
  }
}

__global__ void finalize(const float* accum, float* out) {
  out[0] = accum[0] / accum[1];
}

// ---------------------------------------------------------------------------
extern "C" void kernel_launch(void* const* d_in, const int* in_sizes, int n_in,
                              void* d_out, int out_size, void* d_ws, size_t ws_size,
                              hipStream_t stream) {
  const float* x   = (const float*)d_in[0];   // [B, D]
  const float* L   = (const float*)d_in[1];   // [V, D]
  const int*   tgt = (const int*)  d_in[2];   // [B]
  const float* cw  = (const float*)d_in[3];   // [V]
  float* out = (float*)d_out;

  char* ws = (char*)d_ws;
  const size_t o_xb   = 0;                                        // 16.8 MB
  const size_t o_Lb   = o_xb + (size_t)B_ROWS * D_DIM * 2;        // 131 MB
  const size_t o_part = o_Lb + (size_t)V_DIM * D_DIM * 2;         // 16.4 MB
  const size_t o_gath = o_part + (size_t)B_ROWS * NCHUNK * 8;     // 16 KB
  const size_t o_acc  = o_gath + (size_t)B_ROWS * 4;

  __bf16* xb   = (__bf16*)(ws + o_xb);
  __bf16* Lb   = (__bf16*)(ws + o_Lb);
  float2* part = (float2*)(ws + o_part);
  float*  gath = (float*)(ws + o_gath);
  float*  acc  = (float*)(ws + o_acc);

  // (0) convert x, L to bf16 (bf16 L fits MI455X's 192MB L2 -> GEMM re-reads hit L2)
  {
    int n4 = (B_ROWS * D_DIM) / 4;
    cvt_f32_bf16<<<(n4 + 255) / 256, 256, 0, stream>>>(x, (unsigned short*)xb, n4);
  }
  {
    int n4 = (V_DIM * D_DIM) / 4;
    cvt_f32_bf16<<<(n4 + 255) / 256, 256, 0, stream>>>(L, (unsigned short*)Lb, n4);
  }
  init_accum<<<1, 32, 0, stream>>>(acc);

  // (1) GEMM + fused CE epilogue: grid (V/256, B/128)
  {
    dim3 grid(V_DIM / 256, B_ROWS / 128, 1);
    gemm_ce<<<grid, 256, 0, stream>>>(xb, Lb, tgt, part, gath);
  }

  // (2) per-row merge + weighted NLL accumulate
  reduce_rows<<<B_ROWS / 8, 256, 0, stream>>>(part, gath, tgt, cw, acc);

  // (3) scalar finalize
  finalize<<<1, 1, 0, stream>>>(acc, out);
}
